// DcMoeBlock_47399259079015
// MI455X (gfx1250) — matrix operand order repaired and checked by reading, not compiled, not run on hardware
//
#include <hip/hip_runtime.h>

// Problem constants (match reference)
#define G_   4
#define S_   2048
#define M_   1024
#define E_   8
#define H_   4096
#define K_   2
#define CAP_ 640

typedef __attribute__((ext_vector_type(16))) __bf16 v16bf;
typedef __attribute__((ext_vector_type(8)))  float  v8f;

union Frag { v16bf v; unsigned int u[8]; };

__device__ __forceinline__ unsigned short f2bf(float x) {
  unsigned int u = __float_as_uint(x);
  u += 0x7FFFu + ((u >> 16) & 1u);   // round-to-nearest-even
  return (unsigned short)(u >> 16);
}

// --- CDNA5 async global->LDS copy (ASYNCcnt path), 16B per lane ------------
__device__ __forceinline__ unsigned lds_addr(const void* p) {
  // generic pointer to LDS: low 32 bits are the LDS byte address (ISA 10.2)
  return (unsigned)(unsigned long long)p;
}
__device__ __forceinline__ void async_copy_b128(unsigned lds, const void* gaddr) {
  asm volatile("global_load_async_to_lds_b128 %0, %1, off"
               :: "v"(lds), "v"(gaddr) : "memory");
}
__device__ __forceinline__ void wait_async0() {
  asm volatile("s_wait_asynccnt 0" ::: "memory");
}

// ---------------------------------------------------------------------------
// 1) Router: logits = x @ router_w, softmax, top-2, loss partial sums.
//    sums layout: [0..G*E) = sum_s probs ; [G*E..2GE) = sum_s em ; [2GE] = sum z
// ---------------------------------------------------------------------------
__global__ __launch_bounds__(128) void router_kernel(
    const float* __restrict__ inputs, const float* __restrict__ router_w,
    int* __restrict__ top_i, float* __restrict__ top_p, float* __restrict__ sums) {
  __shared__ float rw[M_ * E_];          // 32 KB
  __shared__ float bsum_p[E_], bsum_em[E_], bz[1];
  int tid = threadIdx.x;
  for (int i = tid; i < M_ * E_; i += 128) rw[i] = router_w[i];
  if (tid < E_) { bsum_p[tid] = 0.f; bsum_em[tid] = 0.f; }
  if (tid == 0) bz[0] = 0.f;
  __syncthreads();

  int wid = tid >> 5, lane = tid & 31;
  int t = blockIdx.x * 4 + wid;          // token index in [0, G*S)
  int g = t / S_;
  const float* x = inputs + (size_t)t * M_;

  float acc[E_] = {0.f,0.f,0.f,0.f,0.f,0.f,0.f,0.f};
  for (int m = lane; m < M_; m += 32) {
    float xv = x[m];
#pragma unroll
    for (int e = 0; e < E_; ++e) acc[e] += xv * rw[m * E_ + e];
  }
#pragma unroll
  for (int off = 16; off > 0; off >>= 1) {
#pragma unroll
    for (int e = 0; e < E_; ++e) acc[e] += __shfl_xor(acc[e], off, 32);
  }

  if (lane == 0) {
    float mx = acc[0];
#pragma unroll
    for (int e = 1; e < E_; ++e) mx = fmaxf(mx, acc[e]);
    float ex[E_], se = 0.f;
#pragma unroll
    for (int e = 0; e < E_; ++e) { ex[e] = __expf(acc[e] - mx); se += ex[e]; }
    float lse = mx + logf(se);
    float p[E_];
#pragma unroll
    for (int e = 0; e < E_; ++e) p[e] = ex[e] / se;
    int i1 = 0;
#pragma unroll
    for (int e = 1; e < E_; ++e) if (p[e] > p[i1]) i1 = e;
    int i2 = (i1 == 0) ? 1 : 0;
#pragma unroll
    for (int e = 0; e < E_; ++e) if (e != i1 && p[e] > p[i2]) i2 = e;
    top_i[t * K_ + 0] = i1; top_i[t * K_ + 1] = i2;
    top_p[t * K_ + 0] = p[i1]; top_p[t * K_ + 1] = p[i2];
#pragma unroll
    for (int e = 0; e < E_; ++e) atomicAdd(&bsum_p[e], p[e]);
    atomicAdd(&bsum_em[i1], 1.f); atomicAdd(&bsum_em[i2], 1.f);
    atomicAdd(&bz[0], lse * lse);
  }
  __syncthreads();
  if (tid < E_) {
    atomicAdd(&sums[g * E_ + tid], bsum_p[tid]);
    atomicAdd(&sums[G_ * E_ + g * E_ + tid], bsum_em[tid]);
  }
  if (tid == 0) atomicAdd(&sums[2 * G_ * E_], bz[0]);
}

// ---------------------------------------------------------------------------
// 2) Capacity scan: per group, wave-parallel prefix over flat (s,k) order.
// ---------------------------------------------------------------------------
__global__ __launch_bounds__(32) void scan_kernel(
    const int* __restrict__ top_i, const float* __restrict__ paddings,
    int* __restrict__ slot_e, int* __restrict__ slot_c, int* __restrict__ expert_src) {
  int g = blockIdx.x, lane = threadIdx.x;
  const int CH = (S_ * K_) / 32;       // 128 entries per lane
  int base = lane * CH;
  int lcnt[E_] = {0,0,0,0,0,0,0,0};
  for (int i = 0; i < CH; ++i) {
    int f = base + i, s = f >> 1;
    if (paddings[g * S_ + s] == 0.f) {
      int e = top_i[(size_t)(g * S_ + s) * K_ + (f & 1)];
      lcnt[e]++;
    }
  }
  int run[E_];
#pragma unroll
  for (int e = 0; e < E_; ++e) {
    int x = lcnt[e];
#pragma unroll
    for (int off = 1; off < 32; off <<= 1) {
      int y = __shfl_up(x, off, 32);
      if (lane >= off) x += y;
    }
    run[e] = x - lcnt[e];              // exclusive prefix = count before my chunk
  }
  for (int i = 0; i < CH; ++i) {
    int f = base + i, s = f >> 1;
    size_t ti = (size_t)(g * S_ + s) * K_ + (f & 1);
    int sc = -1, se = 0;
    if (paddings[g * S_ + s] == 0.f) {
      int e = top_i[ti];
      int c = run[e]++;
      se = e;
      if (c < CAP_) {
        sc = c;
        expert_src[((size_t)e * G_ + g) * CAP_ + c] = s;
      }
    }
    slot_e[ti] = se; slot_c[ti] = sc;
  }
}

// ---------------------------------------------------------------------------
// 3) Gather tokens into expert_in (bf16, row-major [slot][M]).
// ---------------------------------------------------------------------------
__global__ __launch_bounds__(256) void gather_kernel(
    const float* __restrict__ inputs, const int* __restrict__ expert_src,
    unsigned short* __restrict__ Xb) {
  int b = blockIdx.x;                  // (e*G+g)*CAP + c
  int src = expert_src[b];
  if (src < 0) return;                 // buffer pre-zeroed
  int g = (b / CAP_) % G_;
  const float* x = inputs + ((size_t)g * S_ + src) * M_;
  unsigned short* dst = Xb + (size_t)b * M_;
  for (int m = threadIdx.x; m < M_; m += 256) dst[m] = f2bf(x[m]);
}

// ---------------------------------------------------------------------------
// 4) Weight f32 -> bf16 conversion with WMMA-B swizzle.
//    src: [E][Krows][N] f32 (row-major, K-major)
//    dst: [E][Krows/32][N][16] dwords; dword d of row n in panel kk holds
//         bf16 pair (src[kk*32+2d][n], src[kk*32+2d+1][n]) -> exactly the
//         LDS image the WMMA B fragment wants (frag.u[v] = row[(v&4)*2+4*hl+(v&3)]).
//    One thread per (e,kk,n): 32 coalesced f32 reads, 64B dword writes.
// ---------------------------------------------------------------------------
__global__ __launch_bounds__(256) void cvt_swizzle_kernel(
    const float* __restrict__ src, unsigned int* __restrict__ dst,
    int Krows, int N) {
  long tid = (long)blockIdx.x * 256 + threadIdx.x;
  long total = (long)E_ * (Krows / 32) * N;
  if (tid >= total) return;
  int n = (int)(tid % N);
  long t2 = tid / N;
  int kk = (int)(t2 % (Krows / 32));
  int e  = (int)(t2 / (Krows / 32));
  const float* s = src + ((size_t)e * Krows + (size_t)kk * 32) * N + n;
  unsigned int* d = dst + (size_t)tid * 16;
#pragma unroll
  for (int p = 0; p < 16; ++p) {
    unsigned lo = f2bf(s[(size_t)(2 * p) * N]);
    unsigned hi = f2bf(s[(size_t)(2 * p + 1) * N]);
    d[p] = lo | (hi << 16);
  }
}

// ---------------------------------------------------------------------------
// Fragment helpers (ISA 7.12.2 layouts, wave32)
//   A (16x32 bf16): lane = 16*hl + M15; dwords 0-3: K pair (8*hl + 2v),
//                   dwords 4-7: K pair (16 + 8*hl + 2(v-4))
//   B (32x16 bf16): lane = 16*hl + N; dword v: same K pair as A = row[(v&4)*2 + 4*hl + (v&3)]
//   C/D (16x16 f32): elem r -> M = r + 8*hl, N = lane & 15
// ---------------------------------------------------------------------------
#define A_STRIDE_B 80     // A LDS row stride bytes (16B aligned; 20 dwords -> conflict-free)
#define B_STRIDE_B 80     // B LDS row stride bytes ([n][16 kpair dwords] + pad)

__device__ __forceinline__ void load_fragA(Frag& a, const unsigned char* rowbase, int hl) {
  const unsigned int* rowu = (const unsigned int*)rowbase;
#pragma unroll
  for (int v = 0; v < 4; ++v) {
    a.u[v]     = rowu[4 * hl + v];
    a.u[4 + v] = rowu[8 + 4 * hl + v];
  }
}
__device__ __forceinline__ void load_fragB(Frag& b, const unsigned char* rowbase, int hl) {
  const unsigned int* rowu = (const unsigned int*)rowbase;
#pragma unroll
  for (int v = 0; v < 8; ++v) b.u[v] = rowu[(v & 4) * 2 + 4 * hl + (v & 3)];
}

// ---------------------------------------------------------------------------
// 5) FFN1: h = silu(X@Wg) * (X@W0), X:[CAP,M] bf16 row-major,
//    Wg/W0: swizzled bf16 (see cvt_swizzle). Block 256 thr = 8 waves;
//    tile 64(c) x 128(h); 8 WMMAs per K-step; double-buffered async staging.
// ---------------------------------------------------------------------------
__global__ __launch_bounds__(256) void ffn1_kernel(
    const unsigned short* __restrict__ Xb, const unsigned int* __restrict__ Wg,
    const unsigned int* __restrict__ W0, unsigned short* __restrict__ Hout) {
  int eg = blockIdx.z, e = eg / G_;
  int c0 = blockIdx.y * 64, h0 = blockIdx.x * 128;
  const unsigned char* Xg  = (const unsigned char*)(Xb + (size_t)eg * CAP_ * M_);
  const unsigned char* wgg = (const unsigned char*)Wg;
  const unsigned char* w0g = (const unsigned char*)W0;

  __shared__ unsigned char AuL[2][64  * A_STRIDE_B];   // 2 x 5 KB
  __shared__ unsigned char BgL[2][128 * B_STRIDE_B];   // 2 x 10 KB
  __shared__ unsigned char BuL[2][128 * B_STRIDE_B];   // 2 x 10 KB

  int tid = threadIdx.x, lane = tid & 31, w = tid >> 5;
  int wrow = w >> 1, wcol = w & 1;
  int hl = lane >> 4, l15 = lane & 15;

  v8f accg[4], accu[4];
#pragma unroll
  for (int j = 0; j < 4; ++j)
#pragma unroll
    for (int r = 0; r < 8; ++r) { accg[j][r] = 0.f; accu[j][r] = 0.f; }

  auto prefetch = [&](int kk, int buf) {
    {   // A tile: 64 rows x 64B = 256 x B128 chunks (1 per thread)
      int r = tid >> 2, ch = tid & 3;
      async_copy_b128(lds_addr(&AuL[buf][r * A_STRIDE_B + ch * 16]),
                      Xg + ((size_t)(c0 + r) * M_ + (size_t)kk * 32) * 2 + ch * 16);
    }
    // B tiles (swizzled): panel kk, rows n = h0..h0+127, 64B/row = 4 chunks
#pragma unroll
    for (int j = 0; j < 2; ++j) {  // 512 chunks per tile -> 2/thread
      int idx = tid + 256 * j, n = idx >> 2, ch = idx & 3;
      size_t gofs = (((size_t)(e * (M_ / 32) + kk) * H_ + h0 + n) * 64) + ch * 16;
      unsigned lofs = (unsigned)(n * B_STRIDE_B + ch * 16);
      async_copy_b128(lds_addr(&BgL[buf][lofs]), wgg + gofs);
      async_copy_b128(lds_addr(&BuL[buf][lofs]), w0g + gofs);
    }
  };

  prefetch(0, 0);
  for (int kk = 0; kk < M_ / 32; ++kk) {
    int buf = kk & 1;
    wait_async0();            // my wave's copies into buf are done
    __syncthreads();          // everyone's copies done; prior reads of other buf done
    if (kk + 1 < M_ / 32) prefetch(kk + 1, buf ^ 1);

    Frag a;
    load_fragA(a, &AuL[buf][(wrow * 16 + l15) * A_STRIDE_B], hl);
#pragma unroll
    for (int j = 0; j < 4; ++j) {
      int n = wcol * 64 + j * 16 + l15;
      Frag bg, bu;
      load_fragB(bg, &BgL[buf][n * B_STRIDE_B], hl);
      load_fragB(bu, &BuL[buf][n * B_STRIDE_B], hl);
      accg[j] = __builtin_amdgcn_wmma_f32_16x16x32_bf16(false, a.v, false, bg.v,
                                                        (short)0, accg[j], false, false);
      accu[j] = __builtin_amdgcn_wmma_f32_16x16x32_bf16(false, a.v, false, bu.v,
                                                        (short)0, accu[j], false, false);
    }
  }

#pragma unroll
  for (int j = 0; j < 4; ++j) {
    int col = h0 + wcol * 64 + j * 16 + l15;
#pragma unroll
    for (int r = 0; r < 8; ++r) {
      int row = c0 + wrow * 16 + r + 8 * hl;
      float gv = accg[j][r], uv = accu[j][r];
      float hv = (gv / (1.f + __expf(-gv))) * uv;     // silu(g) * u
      Hout[((size_t)eg * CAP_ + row) * H_ + col] = f2bf(hv);
    }
  }
}

// ---------------------------------------------------------------------------
// 6) FFN2: out = h @ wo, h:[CAP,H] bf16 row-major, wo: swizzled bf16 -> f32
// ---------------------------------------------------------------------------
__global__ __launch_bounds__(256) void ffn2_kernel(
    const unsigned short* __restrict__ Hb, const unsigned int* __restrict__ Wo,
    float* __restrict__ Eout) {
  int eg = blockIdx.z, e = eg / G_;
  int c0 = blockIdx.y * 64, m0 = blockIdx.x * 128;
  const unsigned char* Hg  = (const unsigned char*)(Hb + (size_t)eg * CAP_ * H_);
  const unsigned char* wog = (const unsigned char*)Wo;

  __shared__ unsigned char AuL[2][64  * A_STRIDE_B];
  __shared__ unsigned char BsL[2][128 * B_STRIDE_B];

  int tid = threadIdx.x, lane = tid & 31, w = tid >> 5;
  int wrow = w >> 1, wcol = w & 1;
  int hl = lane >> 4, l15 = lane & 15;

  v8f acc[4];
#pragma unroll
  for (int j = 0; j < 4; ++j)
#pragma unroll
    for (int r = 0; r < 8; ++r) acc[j][r] = 0.f;

  auto prefetch = [&](int kk, int buf) {
    {
      int r = tid >> 2, ch = tid & 3;
      async_copy_b128(lds_addr(&AuL[buf][r * A_STRIDE_B + ch * 16]),
                      Hg + ((size_t)(c0 + r) * H_ + (size_t)kk * 32) * 2 + ch * 16);
    }
#pragma unroll
    for (int j = 0; j < 2; ++j) {
      int idx = tid + 256 * j, n = idx >> 2, ch = idx & 3;
      size_t gofs = (((size_t)(e * (H_ / 32) + kk) * M_ + m0 + n) * 64) + ch * 16;
      async_copy_b128(lds_addr(&BsL[buf][n * B_STRIDE_B + ch * 16]), wog + gofs);
    }
  };

  prefetch(0, 0);
  for (int kk = 0; kk < H_ / 32; ++kk) {
    int buf = kk & 1;
    wait_async0();
    __syncthreads();
    if (kk + 1 < H_ / 32) prefetch(kk + 1, buf ^ 1);

    Frag a;
    load_fragA(a, &AuL[buf][(wrow * 16 + l15) * A_STRIDE_B], hl);
#pragma unroll
    for (int j = 0; j < 4; ++j) {
      int n = wcol * 64 + j * 16 + l15;
      Frag b;
      load_fragB(b, &BsL[buf][n * B_STRIDE_B], hl);
      acc[j] = __builtin_amdgcn_wmma_f32_16x16x32_bf16(false, a.v, false, b.v,
                                                       (short)0, acc[j], false, false);
    }
  }

#pragma unroll
  for (int j = 0; j < 4; ++j) {
    int col = m0 + wcol * 64 + j * 16 + l15;
#pragma unroll
    for (int r = 0; r < 8; ++r) {
      int row = c0 + wrow * 16 + r + 8 * hl;
      Eout[((size_t)eg * CAP_ + row) * M_ + col] = acc[j][r];
    }
  }
}

// ---------------------------------------------------------------------------
// 7) Combine: out[g,s,:] = sum_k p_k * expert_out[e_k, g, c_k, :]
// ---------------------------------------------------------------------------
__global__ __launch_bounds__(256) void combine_kernel(
    const float* __restrict__ top_p, const int* __restrict__ slot_e,
    const int* __restrict__ slot_c, const float* __restrict__ Eout,
    float* __restrict__ out) {
  int t = blockIdx.x;                  // g*S + s
  int g = t / S_;
  int e0 = slot_e[(size_t)t * K_ + 0], c0 = slot_c[(size_t)t * K_ + 0];
  int e1 = slot_e[(size_t)t * K_ + 1], c1 = slot_c[(size_t)t * K_ + 1];
  float p0 = top_p[(size_t)t * K_ + 0], p1 = top_p[(size_t)t * K_ + 1];
  const float* x0 = Eout + (((size_t)e0 * G_ + g) * CAP_ + (c0 < 0 ? 0 : c0)) * M_;
  const float* x1 = Eout + (((size_t)e1 * G_ + g) * CAP_ + (c1 < 0 ? 0 : c1)) * M_;
  float* o = out + (size_t)t * M_;
  for (int m = threadIdx.x; m < M_; m += 256) {
    float a = 0.f;
    if (c0 >= 0) a += p0 * x0[m];
    if (c1 >= 0) a += p1 * x1[m];
    o[m] = a;
  }
}

// ---------------------------------------------------------------------------
// 8) Aux loss finalize
// ---------------------------------------------------------------------------
__global__ void finalize_kernel(const float* __restrict__ sums, float* __restrict__ aux) {
  float lb = 0.f;
  for (int i = 0; i < G_ * E_; ++i)
    lb += (sums[G_ * E_ + i] / (float)S_) * (sums[i] / (float)S_);
  lb = lb / (float)(G_ * E_) * (float)(E_ * E_);
  float z = sums[2 * G_ * E_] / (float)(G_ * S_);
  aux[0] = 0.01f * lb + 0.001f * z;
}

// ---------------------------------------------------------------------------
extern "C" void kernel_launch(void* const* d_in, const int* in_sizes, int n_in,
                              void* d_out, int out_size, void* d_ws, size_t ws_size,
                              hipStream_t stream) {
  (void)in_sizes; (void)n_in; (void)out_size; (void)ws_size;
  const float* inputs   = (const float*)d_in[0];
  const float* paddings = (const float*)d_in[1];
  const float* router_w = (const float*)d_in[2];
  const float* wi_gate  = (const float*)d_in[3];
  const float* wi_0     = (const float*)d_in[4];
  const float* wo       = (const float*)d_in[5];
  float* out = (float*)d_out;

  char* ws = (char*)d_ws;
  size_t off = 0;
  auto take = [&](size_t bytes) {
    char* p = ws + off;
    off = (off + bytes + 255) & ~(size_t)255;
    return p;
  };

  int*   top_i      = (int*)  take((size_t)G_ * S_ * K_ * 4);
  float* top_p      = (float*)take((size_t)G_ * S_ * K_ * 4);
  int*   slot_e     = (int*)  take((size_t)G_ * S_ * K_ * 4);
  int*   slot_c     = (int*)  take((size_t)G_ * S_ * K_ * 4);
  int*   expert_src = (int*)  take((size_t)E_ * G_ * CAP_ * 4);
  float* sums       = (float*)take(512);
  unsigned int* wgb = (unsigned int*)take((size_t)E_ * M_ * H_ * 2);   // swizzled bf16
  unsigned int* w0b = (unsigned int*)take((size_t)E_ * M_ * H_ * 2);
  unsigned int* wob = (unsigned int*)take((size_t)E_ * H_ * M_ * 2);
  unsigned short* xb = (unsigned short*)take((size_t)E_ * G_ * CAP_ * M_ * 2);
  unsigned short* hb = (unsigned short*)take((size_t)E_ * G_ * CAP_ * H_ * 2);
  float*        eout = (float*)take((size_t)E_ * G_ * CAP_ * M_ * 4);

  hipMemsetAsync(sums, 0, 512, stream);
  hipMemsetAsync(expert_src, 0xFF, (size_t)E_ * G_ * CAP_ * 4, stream);   // -1
  hipMemsetAsync(xb, 0, (size_t)E_ * G_ * CAP_ * M_ * 2, stream);

  router_kernel<<<(G_ * S_) / 4, 128, 0, stream>>>(inputs, router_w, top_i, top_p, sums);
  scan_kernel<<<G_, 32, 0, stream>>>(top_i, paddings, slot_e, slot_c, expert_src);
  gather_kernel<<<E_ * G_ * CAP_, 256, 0, stream>>>(inputs, expert_src, xb);

  // swizzling f32->bf16 weight conversion: one thread per (e, k-panel, n)
  long n_wi = (long)E_ * (M_ / 32) * H_;   // wi_gate / wi_0
  long n_wo = (long)E_ * (H_ / 32) * M_;   // wo
  cvt_swizzle_kernel<<<(unsigned)((n_wi + 255) / 256), 256, 0, stream>>>(wi_gate, wgb, M_, H_);
  cvt_swizzle_kernel<<<(unsigned)((n_wi + 255) / 256), 256, 0, stream>>>(wi_0,    w0b, M_, H_);
  cvt_swizzle_kernel<<<(unsigned)((n_wo + 255) / 256), 256, 0, stream>>>(wo,      wob, H_, M_);

  ffn1_kernel<<<dim3(H_ / 128, CAP_ / 64, E_ * G_), 256, 0, stream>>>(xb, wgb, w0b, hb);
  ffn2_kernel<<<dim3(M_ / 128, CAP_ / 64, E_ * G_), 256, 0, stream>>>(hb, wob, eout);

  combine_kernel<<<G_ * S_, 256, 0, stream>>>(top_p, slot_e, slot_c, eout, out);
  finalize_kernel<<<1, 1, 0, stream>>>(sums, out + (size_t)G_ * S_ * M_);
}
